// ClusteringLoss_30116310679745
// MI455X (gfx1250) — 8-bit matrix operand order repaired and checked by reading, compiled, not run on hardware
//
#include <hip/hip_runtime.h>

typedef int v8i __attribute__((ext_vector_type(8)));

#define NPTS   4096
#define NCLS   32
#define NEVT   8


__global__ __launch_bounds__(256)
void zero_acc_kernel(int* __restrict__ p, int n) {
    int i = blockIdx.x * blockDim.x + threadIdx.x;
    if (i < n) p[i] = 0;
}

__global__ __launch_bounds__(256)
void argmax_onehot_kernel(const float* __restrict__ pred, unsigned char* __restrict__ LT8) {
    int i = blockIdx.x * blockDim.x + threadIdx.x;
    if (i >= NPTS) return;
    const float4* p4 = (const float4*)(pred + (size_t)i * NCLS);
    float best = -3.402823466e38f;
    int lbl = 0;
    #pragma unroll
    for (int q = 0; q < NCLS / 4; ++q) {
        float4 v = p4[q];
        if (v.x > best) { best = v.x; lbl = 4 * q + 0; }
        if (v.y > best) { best = v.y; lbl = 4 * q + 1; }
        if (v.z > best) { best = v.z; lbl = 4 * q + 2; }
        if (v.w > best) { best = v.w; lbl = 4 * q + 3; }
    }
    #pragma unroll
    for (int l = 0; l < NCLS; ++l)
        LT8[(size_t)l * NPTS + i] = (unsigned char)(l == lbl ? 1 : 0);
}

__global__ __launch_bounds__(256)
void pack_mask_kernel(const float* __restrict__ targ, unsigned char* __restrict__ MT8) {
    int idx = blockIdx.x * blockDim.x + threadIdx.x;
    if (idx >= NEVT * NPTS) return;
    int b = idx >> 12, i = idx & (NPTS - 1);
    const float4* t4 = (const float4*)(targ + (size_t)idx * NCLS);
    unsigned char* dst = MT8 + (size_t)b * NCLS * NPTS + i;
    #pragma unroll
    for (int q = 0; q < NCLS / 4; ++q) {
        float4 v = t4[q];
        dst[(size_t)(4 * q + 0) * NPTS] = (unsigned char)(v.x == 1.0f);
        dst[(size_t)(4 * q + 1) * NPTS] = (unsigned char)(v.y == 1.0f);
        dst[(size_t)(4 * q + 2) * NPTS] = (unsigned char)(v.z == 1.0f);
        dst[(size_t)(4 * q + 3) * NPTS] = (unsigned char)(v.w == 1.0f);
    }
}

__global__ __launch_bounds__(32)
void wmma_count_kernel(const unsigned char* __restrict__ MT8,
                       const unsigned char* __restrict__ LT8,
                       int* __restrict__ acc_n, int* __restrict__ acc_c2) {
    const int b    = blockIdx.x >> 2;
    const int m0   = ((blockIdx.x >> 1) & 1) * 16;
    const int n0   = (blockIdx.x & 1) * 16;
    const int lane = threadIdx.x;
    const int row  = lane & 15;
    const int half = lane >> 4;

    const unsigned char* Abase =
        MT8 + ((size_t)b * NCLS + (size_t)(m0 + row)) * NPTS + half * 8;
    const unsigned char* Bbase =
        LT8 + (size_t)(n0 + row) * NPTS + half * 8;

    v8i c = {};
    for (int i0 = 0; i0 < NPTS; i0 += 64) {
        uint2 a01 = *(const uint2*)(Abase + i0 + 0);
        uint2 a23 = *(const uint2*)(Abase + i0 + 16);
        uint2 a45 = *(const uint2*)(Abase + i0 + 32);
        uint2 a67 = *(const uint2*)(Abase + i0 + 48);
        v8i a;
        a[0] = (int)a01.x; a[1] = (int)a01.y;
        a[2] = (int)a23.x; a[3] = (int)a23.y;
        a[4] = (int)a45.x; a[5] = (int)a45.y;
        a[6] = (int)a67.x; a[7] = (int)a67.y;

        uint2 b01 = *(const uint2*)(Bbase + i0 + 0);
        uint2 b23 = *(const uint2*)(Bbase + i0 + 16);
        uint2 b45 = *(const uint2*)(Bbase + i0 + 32);
        uint2 b67 = *(const uint2*)(Bbase + i0 + 48);
        v8i bm;
        bm[0] = (int)b01.x; bm[1] = (int)b01.y; bm[2] = (int)b23.x; bm[3] = (int)b23.y;
        bm[4] = (int)b45.x; bm[5] = (int)b45.y; bm[6] = (int)b67.x; bm[7] = (int)b67.y;

        c = __builtin_amdgcn_wmma_i32_16x16x64_iu8(false, a, false, bm, c, false, false);
    }

    #pragma unroll
    for (int j = 0; j < 8; ++j) {
        int r = m0 + j + half * 8;
        int v = c[j];
        atomicAdd(&acc_n[b * NCLS + r], v);
        atomicAdd(&acc_c2[b * NCLS + r], v * v);
    }
}

__global__ __launch_bounds__(256)
void finalize_kernel(const int* __restrict__ acc_n, const int* __restrict__ acc_c2,
                     float* __restrict__ out) {
    __shared__ float s_loss[256];
    __shared__ float s_cmp[256];
    int t = threadIdx.x;
    float n  = (float)acc_n[t];
    float s2 = (float)acc_c2[t];
    s_loss[t] = s2 - 0.5f * n * (n + 1.0f);
    s_cmp[t]  = 0.5f * n * (n - 1.0f);
    __syncthreads();
    #pragma unroll
    for (int off = 128; off > 0; off >>= 1) {
        if (t < off) { s_loss[t] += s_loss[t + off]; s_cmp[t] += s_cmp[t + off]; }
        __syncthreads();
    }
    if (t == 0) out[0] = s_loss[0] / s_cmp[0];
}

extern "C" void kernel_launch(void* const* d_in, const int* in_sizes, int n_in,
                              void* d_out, int out_size, void* d_ws, size_t ws_size,
                              hipStream_t stream) {
    const float* pred = (const float*)d_in[0];
    const float* targ = (const float*)d_in[1];
    float* out = (float*)d_out;

    unsigned char* LT8 = (unsigned char*)d_ws;
    unsigned char* MT8 = LT8 + (size_t)NCLS * NPTS;
    int* acc_n  = (int*)(MT8 + (size_t)NEVT * NCLS * NPTS);
    int* acc_c2 = acc_n + NEVT * NCLS;

    zero_acc_kernel<<<2, 256, 0, stream>>>(acc_n, 2 * NEVT * NCLS);
    argmax_onehot_kernel<<<NPTS / 256, 256, 0, stream>>>(pred, LT8);
    pack_mask_kernel<<<(NEVT * NPTS) / 256, 256, 0, stream>>>(targ, MT8);
    wmma_count_kernel<<<NEVT * 4, 32, 0, stream>>>(MT8, LT8, acc_n, acc_c2);
    finalize_kernel<<<1, 256, 0, stream>>>(acc_n, acc_c2, out);
}
